// WindowAttention_21878563405900
// MI455X (gfx1250) — matrix operand order repaired and checked by reading, not compiled, not run on hardware
//
#include <hip/hip_runtime.h>

#define NWIN   16384
#define LEN    49
#define LP     64
#define CH     128
#define HEADS  4
#define HD     32
#define NQKV   384
#define QSCALE 0.17677669529663687f   // 1/sqrt(32)

typedef __attribute__((ext_vector_type(16))) _Float16 v16h;
typedef __attribute__((ext_vector_type(8)))  _Float16 v8h;
typedef __attribute__((ext_vector_type(4)))  _Float16 v4h;
typedef __attribute__((ext_vector_type(8)))  float    v8f;
typedef __attribute__((ext_vector_type(4)))  float    v4f;

#define XREG 8    // (64*128/4) float4 / 256 threads
#define MREG 10   // ceil(49*49 / 256)

struct Smem {
  _Float16 wqkv[NQKV * CH];        // qkv_w transposed: [n][k], f16
  _Float16 wproj[CH * CH];         // proj_w transposed: [n][k], f16
  _Float16 qkv[LP * NQKV];         // per-window q,k: [token][col] (v not stored here)
  _Float16 vT[HEADS * HD * LP];    // V transposed: [h][d][token]
  _Float16 p[HEADS * LP * LP];     // softmax probs: [h][m][n]
  _Float16 xo[LP * CH];            // x (f16) then reused as attention output O
  _Float16 bias[HEADS * LP * LP];  // rel-bias + (-60000 pad for n>=49): [h][m][n]
  float    msk[LEN * LEN];         // per-window mask slice
  float    qkvb[NQKV];
  float    projb[CH];
};

__device__ __forceinline__ v8f wmma16(v16h a, v16h b, v8f c) {
  return __builtin_amdgcn_wmma_f32_16x16x32_f16(false, a, false, b, (short)0, c,
                                                false, false);
}

// A fragment (16x32 f16, M rows = lane%16), src row-major [M][K] stride ld.
__device__ __forceinline__ v16h ldsA(const _Float16* base, int ld, int m0, int k0,
                                     int lane) {
  const _Float16* p = base + (m0 + (lane & 15)) * ld + k0 + ((lane >> 4) << 3);
  v8h lo = *(const v8h*)(p);
  v8h hi = *(const v8h*)(p + 16);
  return __builtin_shufflevector(lo, hi, 0, 1, 2, 3, 4, 5, 6, 7, 8, 9, 10, 11, 12,
                                 13, 14, 15);
}

// B fragment (32x16 f16, N cols = lane%16), src stored [N][K] row-major stride ld.
__device__ __forceinline__ v16h ldsB(const _Float16* base, int ld, int n0, int k0,
                                     int lane) {
  const _Float16* p = base + (n0 + (lane & 15)) * ld + k0 + ((lane >> 4) << 3);
  v8h lo = *(const v8h*)(p);
  v8h hi = *(const v8h*)(p + 16);
  return __builtin_shufflevector(lo, hi, 0, 1, 2, 3, 4, 5, 6, 7, 8, 9, 10, 11, 12,
                                 13, 14, 15);
}

// Prefetch next window's x (float4, non-temporal) and mask slice into registers.
__device__ __forceinline__ void fetch_x(v4f (&xr)[XREG], const float* __restrict__ x,
                                        int w, int tid) {
#pragma unroll
  for (int j = 0; j < XREG; ++j) {
    int i = tid + j * 256;           // i < 2048
    int m = i >> 5, c4 = (i & 31) << 2;
    v4f v = {};
    if (m < LEN)
      v = __builtin_nontemporal_load(
          (const v4f*)(x + (size_t)w * LEN * CH + (size_t)m * CH + c4));
    xr[j] = v;
  }
}

__device__ __forceinline__ void fetch_m(float (&mr)[MREG],
                                        const float* __restrict__ mask, int w,
                                        int tid) {
  int mw = w >> 8;
#pragma unroll
  for (int j = 0; j < MREG; ++j) {
    int i = tid + j * 256;
    mr[j] = (i < LEN * LEN) ? mask[(size_t)mw * LEN * LEN + i] : 0.f;
  }
}

__global__ void __launch_bounds__(256)
win_attn(const float* __restrict__ x, const float* __restrict__ mask,
         const float* __restrict__ qkv_w, const float* __restrict__ qkv_b,
         const float* __restrict__ proj_w, const float* __restrict__ proj_b,
         const float* __restrict__ rel_bias, float* __restrict__ out) {
  extern __shared__ char smem_raw[];
  Smem& s = *reinterpret_cast<Smem*>(smem_raw);
  const int tid  = threadIdx.x;
  const int lane = tid & 31;
  const int wv   = __builtin_amdgcn_readfirstlane(tid >> 5);  // scalar wave id

  // ---- per-block one-time: stage weights + combined bias table into LDS ----
  for (int i = tid; i < CH * NQKV; i += 256) {
    int k = i / NQKV, n = i % NQKV;
    s.wqkv[n * CH + k] = (_Float16)qkv_w[i];
  }
  for (int i = tid; i < CH * CH; i += 256) {
    int k = i / CH, n = i % CH;
    s.wproj[n * CH + k] = (_Float16)proj_w[i];
  }
  for (int i = tid; i < HEADS * LP * LP; i += 256) {
    int h = i >> 12, m = (i >> 6) & 63, n = i & 63;
    float bv;
    if (m < LEN && n < LEN) {
      int di = (m / 7 - n / 7 + 6) * 13 + (m % 7 - n % 7 + 6);
      bv = rel_bias[di * HEADS + h];
    } else {
      bv = (n >= LEN) ? -60000.0f : 0.0f;  // pad keys -> exp()==0
    }
    s.bias[i] = (_Float16)bv;
  }
  for (int i = tid; i < NQKV; i += 256) s.qkvb[i] = qkv_b[i];
  for (int i = tid; i < CH; i += 256) s.projb[i] = proj_b[i];

  const int wstep = gridDim.x;
  int w = blockIdx.x;
  v4f   xr[XREG];
  float mr[MREG];
  if (w < NWIN) { fetch_x(xr, x, w, tid); fetch_m(mr, mask, w, tid); }
  __syncthreads();

  for (; w < NWIN; w += wstep) {
    // ---- stage A: spill prefetched regs to LDS (pure LDS stores) ----
#pragma unroll
    for (int j = 0; j < XREG; ++j) {
      int i = tid + j * 256;
      int m = i >> 5, c4 = (i & 31) << 2;
      v4h hv;
#pragma unroll
      for (int q = 0; q < 4; ++q) hv[q] = (_Float16)xr[j][q];
      *(v4h*)(s.xo + m * CH + c4) = hv;
    }
#pragma unroll
    for (int j = 0; j < MREG; ++j) {
      int i = tid + j * 256;
      if (i < LEN * LEN) s.msk[i] = mr[j];
    }
    __syncthreads();

    // issue next window's global loads now; they stay in flight behind B..E
    int wn = w + wstep;
    if (wn < NWIN) { fetch_x(xr, x, wn, tid); fetch_m(mr, mask, wn, tid); }

    // ---- stage B: QKV GEMM [64x128]x[128x384] (96 tiles, K=128) ----
    for (int t = wv; t < 96; t += 8) {
      int mt = t & 3, nt = t >> 2;
      v16h a[4], b[4];
#pragma unroll
      for (int kk = 0; kk < 4; ++kk) {
        a[kk] = ldsA(s.xo, CH, mt * 16, kk * 32, lane);
        b[kk] = ldsB(s.wqkv, CH, nt * 16, kk * 32, lane);
      }
      v8f acc = {};
#pragma unroll
      for (int kk = 0; kk < 4; ++kk) acc = wmma16(a[kk], b[kk], acc);

      int n  = nt * 16 + (lane & 15);
      int mb = mt * 16 + ((lane >> 4) << 3);
      float bv = s.qkvb[n];
      _Float16 hv[8];
#pragma unroll
      for (int r = 0; r < 8; ++r) hv[r] = (_Float16)(acc[r] + bv);
      if (nt < 16) {  // q,k tiles: row-major [token][col]
#pragma unroll
        for (int r = 0; r < 8; ++r) s.qkv[(mb + r) * NQKV + n] = hv[r];
      } else {        // v tiles: transposed only (contiguous -> ds_store_b128)
#pragma unroll
        for (int r = 0; r < 8; ++r) s.vT[(n - 256) * LP + (mb + r)] = hv[r];
      }
    }
    __syncthreads();

    // ---- stage C: S = scale*(Q K^T) + bias + mask ; softmax -> P (f16) ----
    for (int pp = wv; pp < 16; pp += 8) {
      int h = pp >> 2, mt = pp & 3;
      v16h aq = ldsA(s.qkv, NQKV, mt * 16, h * HD, lane);  // q fragment, K=hd=32
      v16h bk[4];
#pragma unroll
      for (int nt = 0; nt < 4; ++nt)
        bk[nt] = ldsB(s.qkv, NQKV, nt * 16, CH + h * HD, lane);
      v8f acc[4];
#pragma unroll
      for (int nt = 0; nt < 4; ++nt) {
        v8f z = {};
        acc[nt] = wmma16(aq, bk[nt], z);
      }
      int mb = mt * 16 + ((lane >> 4) << 3);
      int nl = lane & 15;
#pragma unroll
      for (int r = 0; r < 8; ++r) {
        int m = mb + r;
        float vals[4];
        float mx = -3.0e38f;
#pragma unroll
        for (int nt = 0; nt < 4; ++nt) {
          int n = nt * 16 + nl;
          float sv = acc[nt][r] * QSCALE + (float)s.bias[((h * LP) + m) * LP + n];
          if (m < LEN && n < LEN) sv += s.msk[m * LEN + n];
          vals[nt] = sv;
          mx = fmaxf(mx, sv);
        }
#pragma unroll
        for (int off = 1; off < 16; off <<= 1) mx = fmaxf(mx, __shfl_xor(mx, off));
        float sum = 0.f;
#pragma unroll
        for (int nt = 0; nt < 4; ++nt) {
          vals[nt] = __expf(vals[nt] - mx);
          sum += vals[nt];
        }
#pragma unroll
        for (int off = 1; off < 16; off <<= 1) sum += __shfl_xor(sum, off);
        float inv = 1.f / sum;
#pragma unroll
        for (int nt = 0; nt < 4; ++nt)
          s.p[(h * LP + m) * LP + nt * 16 + nl] = (_Float16)(vals[nt] * inv);
      }
    }
    __syncthreads();

    // ---- stage D: O = P V (per head [64x64]x[64x32]) -> xo (f16) ----
    for (int t = wv; t < 32; t += 8) {
      int mt = t & 3, nt = t >> 2;
      int h = nt >> 1;
      v16h a[2], b[2];
#pragma unroll
      for (int kk = 0; kk < 2; ++kk) {
        a[kk] = ldsA(s.p + h * LP * LP, LP, mt * 16, kk * 32, lane);
        b[kk] = ldsB(s.vT + h * HD * LP, LP, (nt & 1) * 16, kk * 32, lane);
      }
      v8f acc = {};
#pragma unroll
      for (int kk = 0; kk < 2; ++kk) acc = wmma16(a[kk], b[kk], acc);
      int n  = nt * 16 + (lane & 15);
      int mb = mt * 16 + ((lane >> 4) << 3);
#pragma unroll
      for (int r = 0; r < 8; ++r) s.xo[(mb + r) * CH + n] = (_Float16)acc[r];
    }
    __syncthreads();

    // ---- stage E: out = O @ proj_w + proj_b ; write valid 49 rows (NT) ----
    for (int t = wv; t < 32; t += 8) {
      int mt = t & 3, nt = t >> 2;
      v16h a[4], b[4];
#pragma unroll
      for (int kk = 0; kk < 4; ++kk) {
        a[kk] = ldsA(s.xo, CH, mt * 16, kk * 32, lane);
        b[kk] = ldsB(s.wproj, CH, nt * 16, kk * 32, lane);
      }
      v8f acc = {};
#pragma unroll
      for (int kk = 0; kk < 4; ++kk) acc = wmma16(a[kk], b[kk], acc);
      int n  = nt * 16 + (lane & 15);
      int mb = mt * 16 + ((lane >> 4) << 3);
      float pb = s.projb[n];
#pragma unroll
      for (int r = 0; r < 8; ++r) {
        int m = mb + r;
        if (m < LEN)
          __builtin_nontemporal_store(
              acc[r] + pb, out + (size_t)w * LEN * CH + (size_t)m * CH + n);
      }
    }
    __syncthreads();
  }
}

extern "C" void kernel_launch(void* const* d_in, const int* in_sizes, int n_in,
                              void* d_out, int out_size, void* d_ws, size_t ws_size,
                              hipStream_t stream) {
  (void)in_sizes; (void)n_in; (void)out_size; (void)d_ws; (void)ws_size;
  const float* x        = (const float*)d_in[0];
  const float* mask     = (const float*)d_in[1];
  const float* qkv_w    = (const float*)d_in[2];
  const float* qkv_b    = (const float*)d_in[3];
  const float* proj_w   = (const float*)d_in[4];
  const float* proj_b   = (const float*)d_in[5];
  const float* rel_bias = (const float*)d_in[6];
  float* out = (float*)d_out;

  // 2048 blocks x 256 threads (8 wave32 each); grid-stride over 16384 windows
  // dynamic LDS = sizeof(Smem) ~ 290 KB (< 320 KB/WGP)
  win_attn<<<dim3(2048), dim3(256), sizeof(Smem), stream>>>(
      x, mask, qkv_w, qkv_b, proj_w, proj_b, rel_bias, out);
}
